// TTS_1915555414713
// MI455X (gfx1250) — compile-verified, matrix operand order checked by reading, not run on hardware
//
#include <hip/hip_runtime.h>
#include <hip/hip_bf16.h>

typedef __attribute__((ext_vector_type(16))) _Float16 v16h;
typedef __attribute__((ext_vector_type(8)))  float    v8f;

// Problem constants (match reference)
constexpr int kB     = 512;
constexpr int kT     = 256;   // T_ENC
constexpr int kD     = 256;   // ENC_D
constexpr int kH     = 256;
constexpr int kMEL   = 80;
constexpr int kR     = 5;
constexpr int kSTEPS = 40;
constexpr int kS     = 10;    // STYLES
constexpr int kSD    = 256;

constexpr int ROWS = 16;          // batch rows per workgroup
constexpr int TPB  = 512;         // 16 waves (wave32) in the decoder
constexpr int PTPB = 256;         // prep kernels
constexpr int CS   = 512;         // fixed C stride in LDS

// output layout (floats, concatenated flat in return order)
constexpr size_t MEL_BASE    = 0;
constexpr size_t ALPHA_BASE  = (size_t)kB * kSTEPS * kR * kMEL;          // 8,192,000
constexpr size_t ALPHAS_BASE = ALPHA_BASE + (size_t)kB * kSTEPS * kT;    // +5,242,880
constexpr size_t WGT_BASE    = ALPHAS_BASE + (size_t)kB * kSTEPS * kS;   // +204,800

enum { ACT_NONE = 0, ACT_RELU, ACT_SIG, ACT_TANH };

// ---------------------------------------------------------------------------
// Prep kernels
// ---------------------------------------------------------------------------

// dst[n][k] (f16, K padded to Kp with zeros) = src[k][n] (f32)
__global__ void k_transpose_f16(const float* __restrict__ src,
                                _Float16* __restrict__ dst,
                                int K, int N, int Kp) {
    int idx = blockIdx.x * PTPB + threadIdx.x;
    int total = N * Kp;
    if (idx >= total) return;
    int n = idx / Kp, k = idx - n * Kp;
    dst[idx] = (k < K) ? (_Float16)src[(size_t)k * N + n] : (_Float16)0.0f;
}

__global__ void k_f32_to_f16(const float* __restrict__ src,
                             _Float16* __restrict__ dst, size_t n) {
    size_t i = (size_t)blockIdx.x * PTPB + threadIdx.x;
    if (i < n) dst[i] = (_Float16)src[i];
}

// keys_f16[M=131072][256] = mem_f16[M][256] @ WmemT^T  (WMMA, one 16x16 tile/wave)
__global__ void k_keys(const _Float16* __restrict__ memh,
                       const _Float16* __restrict__ WmemT,
                       _Float16* __restrict__ keys) {
    const int wave = threadIdx.x >> 5, lane = threadIdx.x & 31;
    const int r16 = lane & 15, hi = lane >> 4, kh = hi * 16;
    const long tile = (long)blockIdx.x * 8 + wave;   // 131072 tiles total
    const int mt = (int)(tile & 8191);               // 8192 m-tiles
    const int nt = (int)(tile >> 13);                // 16 n-tiles
    v8f acc = {};
    const _Float16* arow = memh  + (size_t)(mt * 16 + r16) * 256;
    const _Float16* brow = WmemT + (size_t)(nt * 16 + r16) * 256;
    for (int kt = 0; kt < 256; kt += 32) {
        v16h a = *(const v16h*)(arow + kt + kh);
        v16h b = *(const v16h*)(brow + kt + kh);
        acc = __builtin_amdgcn_wmma_f32_16x16x32_f16(false, a, false, b,
                                                     (short)0, acc, false, false);
    }
#pragma unroll
    for (int r = 0; r < 8; ++r) {
        int row = mt * 16 + r + 8 * hi;
        int col = nt * 16 + r16;
        keys[(size_t)row * 256 + col] = (_Float16)acc[r];
    }
}

// keys_s[10][256] = style[10][256] @ Wmem_s[256][256]   (tiny, f32)
__global__ void k_keys_s(const float* __restrict__ style,
                         const float* __restrict__ Wmem_s,
                         float* __restrict__ ks) {
    int s = blockIdx.x, h = threadIdx.x;
    float acc = 0.f;
    for (int d = 0; d < kSD; ++d) acc += style[s * kSD + d] * Wmem_s[d * kH + h];
    ks[s * kH + h] = acc;
}

// ---------------------------------------------------------------------------
// Decode kernel: 32 workgroups x 512 threads; each WG owns 16 batch rows.
// ---------------------------------------------------------------------------

struct DecodeArgs {
    const _Float16* memh;   // [B*T][D] f16
    const _Float16* keysh;  // [B*T][H] f16
    const float* keys_s;    // [10][256]
    const int*   inp_mask;
    const float* style;     // [10][256]
    const float* v_att;     // [256]
    const float* v_s;       // [256]
    const float* ww;        // [256]
    const float* wb;        // [1]
    const float *pb1, *pb2, *att_gb, *att_cb, *ab;
    const float *g1_gb, *g1_cb, *g2_gb, *g2_cb, *ob;
    const _Float16 *pw1T, *pw2T, *agkT, *ackT, *WqT, *WqsT;
    const _Float16 *aWT, *g1gkT, *g1ckT, *g2gkT, *g2ckT, *oWT;
    float* state;           // per-WG scratch (floats)
    float* out;             // d_out
};

constexpr int WG_STRIDE = 1280 + 8 * 4096 + 2048;   // indic + 8 bufs + pbuf

// ---- device helpers -------------------------------------------------------

// [16,Kp] f16 A-tile @ [N,Kp] f16 weights -> sC[16][CS] f32 (+bias, activation)
__device__ __forceinline__ void wg_gemm(const _Float16* sA, int Kp,
                                        const _Float16* __restrict__ Bt,
                                        float* sC, int N,
                                        const float* __restrict__ bias, int act) {
    const int wave = threadIdx.x >> 5, lane = threadIdx.x & 31;
    const int r16 = lane & 15, hi = lane >> 4, kh = hi * 16;
    const int ntiles = N >> 4;
    const _Float16* arow = sA + r16 * Kp;
    for (int nt = wave; nt < ntiles; nt += (TPB / 32)) {
        const int ncol = nt * 16 + r16;
        const _Float16* brow = Bt + (size_t)ncol * Kp;
        v8f acc = {};
        for (int kt = 0; kt < Kp; kt += 32) {
            v16h a = *(const v16h*)(arow + kt + kh);
            v16h b = *(const v16h*)(brow + kt + kh);
            acc = __builtin_amdgcn_wmma_f32_16x16x32_f16(false, a, false, b,
                                                         (short)0, acc, false, false);
        }
        float bv = bias ? bias[ncol] : 0.0f;
#pragma unroll
        for (int r = 0; r < 8; ++r) {
            float v = acc[r] + bv;
            if (act == ACT_RELU)      v = fmaxf(v, 0.0f);
            else if (act == ACT_SIG)  v = 1.0f / (1.0f + __expf(-v));
            else if (act == ACT_TANH) v = tanhf(v);
            sC[(r + 8 * hi) * CS + ncol] = v;
        }
    }
}

// sA[16][Kp] = f16([a | b | 0])
__device__ __forceinline__ void fill_A_cat(_Float16* sA, int Kp,
                                           const float* a, int wa,
                                           const float* b, int wb) {
    for (int idx = threadIdx.x; idx < 16 * Kp; idx += TPB) {
        int row = idx / Kp, k = idx - row * Kp;
        float v = 0.f;
        if (k < wa)            v = a[row * wa + k];
        else if (k < wa + wb)  v = b[row * wb + (k - wa)];
        sA[idx] = (_Float16)v;
    }
}

// sA[16][Kp] = f16(sC[:, :Kp])
__device__ __forceinline__ void fill_A_fromC(_Float16* sA, int Kp, const float* sC) {
    for (int idx = threadIdx.x; idx < 16 * Kp; idx += TPB) {
        int row = idx / Kp, k = idx - row * Kp;
        sA[idx] = (_Float16)sC[row * CS + k];
    }
}

// build candidate input [x | r*h] and save u (gate halves live in sC[16][512])
__device__ __forceinline__ void gru_mid(_Float16* sA, int Kp,
                                        const float* x, int wx,
                                        const float* sC, const float* h,
                                        float* u) {
    for (int idx = threadIdx.x; idx < 16 * Kp; idx += TPB) {
        int row = idx / Kp, k = idx - row * Kp;
        float v;
        if (k < wx) v = x[row * wx + k];
        else { int hh = k - wx; v = sC[row * CS + hh] * h[row * 256 + hh]; }
        sA[idx] = (_Float16)v;
    }
    for (int idx = threadIdx.x; idx < 16 * 256; idx += TPB) {
        int row = idx >> 8, hh = idx & 255;
        u[idx] = sC[row * CS + 256 + hh];
    }
}

// h = u*h + (1-u)*c ; optionally o = x + h_new (residual)
__device__ __forceinline__ void gru_combine(const float* u, const float* sC,
                                            float* h, float* o, const float* x) {
    for (int idx = threadIdx.x; idx < 4096; idx += TPB) {
        int row = idx >> 8, hh = idx & 255;
        float uu = u[idx];
        float hn = uu * h[idx] + (1.f - uu) * sC[row * CS + hh];
        h[idx] = hn;
        if (o) o[idx] = x[idx] + hn;
    }
}

__device__ __forceinline__ void save_C(const float* sC, float* dst, int w) {
    for (int idx = threadIdx.x; idx < 16 * w; idx += TPB) {
        int row = idx / w, c = idx - row * w;
        dst[idx] = sC[row * CS + c];
    }
}

// ---- the decoder ----------------------------------------------------------

__global__ void __launch_bounds__(TPB, 1) k_decode(DecodeArgs A) {
    __shared__ __align__(32) _Float16 sA[16 * 512];   // 16 KB
    __shared__ float sC[16 * CS];                     // 32 KB
    __shared__ float sAlphaS[16 * 16];                // 1 KB
    __shared__ float sWgt[16];

    const int tid = threadIdx.x;
    const int wg  = blockIdx.x;
    const int b0  = wg * ROWS;

    float* st    = A.state + (size_t)wg * WG_STRIDE;
    float* indic = st;               // 16*80
    float* ha    = st + 1280;        // 16*256
    float* h1    = ha + 4096;
    float* h2    = h1 + 4096;
    float* ubuf  = h2 + 4096;
    float* xbuf  = ubuf + 4096;      // aco_in, later o2
    float* o1buf = xbuf + 4096;
    float* ctx   = o1buf + 4096;     // later wc
    float* alpha = ctx + 4096;
    float* pbuf  = alpha + 4096;     // 16*128

    // zero initial state (indic, ha, h1, h2 are contiguous)
    for (int i = tid; i < 1280 + 3 * 4096; i += TPB) st[i] = 0.f;
    __syncthreads();

    for (int step = 0; step < kSTEPS; ++step) {
        // ---------- prenet ----------
        fill_A_cat(sA, 96, indic, 80, nullptr, 0);
        __syncthreads();
        wg_gemm(sA, 96, A.pw1T, sC, 256, A.pb1, ACT_RELU);
        __syncthreads();
        fill_A_fromC(sA, 256, sC);
        __syncthreads();
        wg_gemm(sA, 256, A.pw2T, sC, 128, A.pb2, ACT_RELU);
        __syncthreads();
        save_C(sC, pbuf, 128);
        __syncthreads();

        // ---------- attention GRU ----------
        fill_A_cat(sA, 384, pbuf, 128, ha, 256);
        __syncthreads();
        wg_gemm(sA, 384, A.agkT, sC, 512, A.att_gb, ACT_SIG);
        __syncthreads();
        gru_mid(sA, 384, pbuf, 128, sC, ha, ubuf);
        __syncthreads();
        wg_gemm(sA, 384, A.ackT, sC, 256, A.att_cb, ACT_TANH);
        __syncthreads();
        gru_combine(ubuf, sC, ha, nullptr, nullptr);
        __syncthreads();

        // ---------- q = ha @ Wq ----------
        fill_A_cat(sA, 256, ha, 256, nullptr, 0);
        __syncthreads();
        wg_gemm(sA, 256, A.WqT, sC, 256, nullptr, ACT_NONE);
        __syncthreads();

        // ---------- content attention: one wave per batch row ----------
        {
            const int brow = tid >> 5;            // wave id = row (0..15)
            const int lane = tid & 31;
            const int tbase = lane * 8;           // 8 time-steps per lane
            const int gb = b0 + brow;
            const int lim = A.inp_mask[gb];
            const float* qrow = &sC[brow * CS];
            float e[8];
            for (int j = 0; j < 8; ++j) {
                const int t = tbase + j;
                const _Float16* kr = A.keysh + ((size_t)gb * kT + t) * kH;
                __builtin_prefetch(kr + kH, 0, 3);          // next row -> L0/L2
                const v16h* kv = (const v16h*)kr;           // 16 x 16 halves
                float acc = 0.f;
                for (int c = 0; c < 16; ++c) {
                    v16h kk = kv[c];
#pragma unroll
                    for (int u = 0; u < 16; ++u)
                        acc += A.v_att[c * 16 + u] *
                               tanhf((float)kk[u] + qrow[c * 16 + u]);
                }
                e[j] = (t < lim) ? acc : -1e9f;
            }
            float m = -3.4e38f;
            for (int j = 0; j < 8; ++j) m = fmaxf(m, e[j]);
            for (int msk = 1; msk < 32; msk <<= 1) m = fmaxf(m, __shfl_xor(m, msk));
            float ssum = 0.f;
            for (int j = 0; j < 8; ++j) { e[j] = __expf(e[j] - m); ssum += e[j]; }
            for (int msk = 1; msk < 32; msk <<= 1) ssum += __shfl_xor(ssum, msk);
            float inv = 1.0f / ssum;
            for (int j = 0; j < 8; ++j) {
                float a_ = e[j] * inv;
                alpha[brow * 256 + tbase + j] = a_;
                A.out[ALPHA_BASE + (size_t)gb * (kSTEPS * kT) + (size_t)step * kT
                      + tbase + j] = a_;
            }
        }
        __syncthreads();

        // ctx[b'][d] = sum_t alpha * memory  (f16 memory, L2-resident)
        {
            const int rh = tid >> 8;              // 0..1 : two rows in flight
            const int d  = tid & 255;
            for (int brow = rh; brow < 16; brow += 2) {
                const _Float16* mr = A.memh + ((size_t)(b0 + brow) * kT) * kD + d;
                const float* ar = &alpha[brow * 256];
                float acc = 0.f;
                for (int t = 0; t < kT; ++t) acc += ar[t] * (float)mr[(size_t)t * kD];
                ctx[brow * 256 + d] = acc;
            }
        }
        __syncthreads();

        // ---------- style attention (q_s reuses sA = ha) ----------
        wg_gemm(sA, 256, A.WqsT, sC, 256, nullptr, ACT_NONE);
        __syncthreads();
        if (tid < 256) {
            int brow = tid >> 4, ss = tid & 15;
            float e = -1e30f;
            if (ss < kS) {
                const float* kr = A.keys_s + ss * kH;
                const float* qrow = &sC[brow * CS];
                float acc = 0.f;
                for (int h = 0; h < kH; ++h)
                    acc += A.v_s[h] * tanhf(kr[h] + qrow[h]);
                e = acc;
            }
            float m = e;
            for (int msk = 1; msk < 16; msk <<= 1) m = fmaxf(m, __shfl_xor(m, msk));
            float ex = (ss < kS) ? __expf(e - m) : 0.f;
            float sum = ex;
            for (int msk = 1; msk < 16; msk <<= 1) sum += __shfl_xor(sum, msk);
            float a_ = ex / sum;
            sAlphaS[brow * 16 + ss] = a_;
            if (ss < kS)
                A.out[ALPHAS_BASE + (size_t)(b0 + brow) * (kSTEPS * kS)
                      + (size_t)step * kS + ss] = a_;
        }
        if (tid < 16) {
            float acc = 0.f;
            for (int h = 0; h < kH; ++h) acc += ha[tid * 256 + h] * A.ww[h];
            float w = 1.f / (1.f + __expf(-(acc + A.wb[0])));
            sWgt[tid] = w;
            A.out[WGT_BASE + (size_t)(b0 + tid) * kSTEPS + step] = w;
        }
        __syncthreads();

        // wc = ctx + wgt * tanh(ctx_s)
        for (int idx = tid; idx < 4096; idx += TPB) {
            int row = idx >> 8, d = idx & 255;
            float cs = 0.f;
            for (int s2 = 0; s2 < kS; ++s2)
                cs += sAlphaS[row * 16 + s2] * A.style[s2 * kSD + d];
            ctx[idx] = ctx[idx] + sWgt[row] * tanhf(cs);
        }
        __syncthreads();

        // ---------- aco_in = [ha | wc] @ aW + ab ----------
        fill_A_cat(sA, 512, ha, 256, ctx, 256);
        __syncthreads();
        wg_gemm(sA, 512, A.aWT, sC, 256, A.ab, ACT_NONE);
        __syncthreads();
        save_C(sC, xbuf, 256);
        __syncthreads();

        // ---------- GRU1 + residual ----------
        fill_A_cat(sA, 512, xbuf, 256, h1, 256);
        __syncthreads();
        wg_gemm(sA, 512, A.g1gkT, sC, 512, A.g1_gb, ACT_SIG);
        __syncthreads();
        gru_mid(sA, 512, xbuf, 256, sC, h1, ubuf);
        __syncthreads();
        wg_gemm(sA, 512, A.g1ckT, sC, 256, A.g1_cb, ACT_TANH);
        __syncthreads();
        gru_combine(ubuf, sC, h1, o1buf, xbuf);      // o1 = aco_in + h1n
        __syncthreads();

        // ---------- GRU2 + residual ----------
        fill_A_cat(sA, 512, o1buf, 256, h2, 256);
        __syncthreads();
        wg_gemm(sA, 512, A.g2gkT, sC, 512, A.g2_gb, ACT_SIG);
        __syncthreads();
        gru_mid(sA, 512, o1buf, 256, sC, h2, ubuf);
        __syncthreads();
        wg_gemm(sA, 512, A.g2ckT, sC, 256, A.g2_cb, ACT_TANH);
        __syncthreads();
        gru_combine(ubuf, sC, h2, xbuf, o1buf);      // o2 = o1 + h2n (into xbuf)
        __syncthreads();

        // ---------- dense out ----------
        fill_A_cat(sA, 256, xbuf, 256, nullptr, 0);
        __syncthreads();
        wg_gemm(sA, 256, A.oWT, sC, 400, A.ob, ACT_NONE);
        __syncthreads();
        for (int idx = tid; idx < 16 * 400; idx += TPB) {
            int row = idx / 400, c = idx - row * 400;
            float v = sC[row * CS + c];
            A.out[MEL_BASE + (size_t)(b0 + row) * (kSTEPS * kR * kMEL)
                  + (size_t)step * (kR * kMEL) + c] = v;
            if (c >= 320) indic[row * 80 + (c - 320)] = v;   // last frame
        }
        __syncthreads();
    }
}

// ---------------------------------------------------------------------------
// Host launcher
// ---------------------------------------------------------------------------

extern "C" void kernel_launch(void* const* d_in, const int* in_sizes, int n_in,
                              void* d_out, int out_size, void* d_ws, size_t ws_size,
                              hipStream_t stream) {
    (void)in_sizes; (void)n_in; (void)out_size; (void)ws_size;

    const float* memory   = (const float*)d_in[0];
    const int*   inp_mask = (const int*)  d_in[1];
    const float* style    = (const float*)d_in[2];
    const float* pw1 = (const float*)d_in[3],  *pb1 = (const float*)d_in[4];
    const float* pw2 = (const float*)d_in[5],  *pb2 = (const float*)d_in[6];
    const float* att_gk = (const float*)d_in[7],  *att_gb = (const float*)d_in[8];
    const float* att_ck = (const float*)d_in[9],  *att_cb = (const float*)d_in[10];
    const float* Wmem = (const float*)d_in[11];
    const float* Wq   = (const float*)d_in[12];
    const float* v_att = (const float*)d_in[13];
    const float* Wmem_s = (const float*)d_in[14];
    const float* Wq_s   = (const float*)d_in[15];
    const float* v_s    = (const float*)d_in[16];
    const float* ww = (const float*)d_in[17], *wb = (const float*)d_in[18];
    const float* aW = (const float*)d_in[19], *ab = (const float*)d_in[20];
    const float* g1_gk = (const float*)d_in[21], *g1_gb = (const float*)d_in[22];
    const float* g1_ck = (const float*)d_in[23], *g1_cb = (const float*)d_in[24];
    const float* g2_gk = (const float*)d_in[25], *g2_gb = (const float*)d_in[26];
    const float* g2_ck = (const float*)d_in[27], *g2_cb = (const float*)d_in[28];
    const float* oW = (const float*)d_in[29], *ob = (const float*)d_in[30];

    char* ws = (char*)d_ws;
    size_t off = 0;
    auto alloc = [&](size_t bytes) {
        size_t o = off;
        off = (off + bytes + 255) & ~(size_t)255;
        return o;
    };

    const size_t nMem = (size_t)kB * kT * kD;           // 33,554,432
    _Float16* memh  = (_Float16*)(ws + alloc(nMem * 2));
    _Float16* keysh = (_Float16*)(ws + alloc(nMem * 2));

    auto allocW = [&](int N, int Kp) { return (_Float16*)(ws + alloc((size_t)N * Kp * 2)); };
    _Float16* pw1T   = allocW(256, 96);
    _Float16* pw2T   = allocW(128, 256);
    _Float16* agkT   = allocW(512, 384);
    _Float16* ackT   = allocW(256, 384);
    _Float16* WqT    = allocW(256, 256);
    _Float16* WqsT   = allocW(256, 256);
    _Float16* WmemT  = allocW(256, 256);
    _Float16* aWT    = allocW(256, 512);
    _Float16* g1gkT  = allocW(512, 512);
    _Float16* g1ckT  = allocW(256, 512);
    _Float16* g2gkT  = allocW(512, 512);
    _Float16* g2ckT  = allocW(256, 512);
    _Float16* oWT    = allocW(400, 256);
    float* keys_s    = (float*)(ws + alloc((size_t)kS * kH * 4));
    float* state     = (float*)(ws + alloc((size_t)(kB / ROWS) * WG_STRIDE * 4));

    auto launchT = [&](const float* src, _Float16* dst, int K, int N, int Kp) {
        int total = N * Kp;
        k_transpose_f16<<<(total + PTPB - 1) / PTPB, PTPB, 0, stream>>>(src, dst, K, N, Kp);
    };
    launchT(pw1,    pw1T,  80,  256, 96);
    launchT(pw2,    pw2T,  256, 128, 256);
    launchT(att_gk, agkT,  384, 512, 384);
    launchT(att_ck, ackT,  384, 256, 384);
    launchT(Wq,     WqT,   256, 256, 256);
    launchT(Wq_s,   WqsT,  256, 256, 256);
    launchT(Wmem,   WmemT, 256, 256, 256);
    launchT(aW,     aWT,   512, 256, 512);
    launchT(g1_gk,  g1gkT, 512, 512, 512);
    launchT(g1_ck,  g1ckT, 512, 256, 512);
    launchT(g2_gk,  g2gkT, 512, 512, 512);
    launchT(g2_ck,  g2ckT, 512, 256, 512);
    launchT(oW,     oWT,   256, 400, 256);

    k_f32_to_f16<<<(int)((nMem + PTPB - 1) / PTPB), PTPB, 0, stream>>>(memory, memh, nMem);
    k_keys<<<16384, PTPB, 0, stream>>>(memh, WmemT, keysh);
    k_keys_s<<<kS, kH, 0, stream>>>(style, Wmem_s, keys_s);

    DecodeArgs A;
    A.memh = memh; A.keysh = keysh; A.keys_s = keys_s;
    A.inp_mask = inp_mask; A.style = style;
    A.v_att = v_att; A.v_s = v_s; A.ww = ww; A.wb = wb;
    A.pb1 = pb1; A.pb2 = pb2; A.att_gb = att_gb; A.att_cb = att_cb; A.ab = ab;
    A.g1_gb = g1_gb; A.g1_cb = g1_cb; A.g2_gb = g2_gb; A.g2_cb = g2_cb; A.ob = ob;
    A.pw1T = pw1T; A.pw2T = pw2T; A.agkT = agkT; A.ackT = ackT;
    A.WqT = WqT; A.WqsT = WqsT; A.aWT = aWT;
    A.g1gkT = g1gkT; A.g1ckT = g1ckT; A.g2gkT = g2gkT; A.g2ckT = g2ckT; A.oWT = oWT;
    A.state = state; A.out = (float*)d_out;

    k_decode<<<kB / ROWS, TPB, 0, stream>>>(A);
}
